// DataAugmentContrastLoss_1236950581701
// MI455X (gfx1250) — compile-verified, matrix operand order checked by reading, not run on hardware
//
#include <hip/hip_runtime.h>
#include <hip/hip_bf16.h>

// ---------------------------------------------------------------------------
// DataAugmentContrastLoss on MI455X (gfx1250, wave32, WMMA + async-to-LDS)
//   B = 4096 anchors, C = 256 features, T = 0.1
//   loss = mean_i [ logsumexp_j( [oo_masked | oa] ) - oa_diag_i ]
//
// Pipeline:
//   1) normalize rows -> f16 workspace copies (o, a)
//   2) zero per-row partial accumulators + output
//   3) fused GEMM+LSE: 16x16 WMMA tiles; key tiles double-buffered in LDS via
//      global_load_async_to_lds_b128 (ASYNCcnt), shared by 4 waves per block;
//      fixed-shift exp-sums are additive -> per-row atomic partials
//   4) finalize: lse = 10 + log(s), loss = mean(lse - pos)
// ---------------------------------------------------------------------------

#define BATCH 4096
#define CHAN  256
#define INV_T 10.0f
#define LSE_SHIFT 10.0f     // |dot/T| <= 10 for unit vectors: safe fixed shift

#define TILE_BYTES 8192     // 16 key rows * 512 B (f16) per staged tile

typedef __attribute__((ext_vector_type(16))) _Float16 v16h;
typedef __attribute__((ext_vector_type(8)))  _Float16 v8h;
typedef __attribute__((ext_vector_type(8)))  float    v8f;

// ---------------------------------------------------------------------------
// Kernel 1: L2-normalize rows of both feature matrices, emit f16.
// One wave (32 lanes) per row; each lane handles 8 contiguous floats.
// ---------------------------------------------------------------------------
__global__ __launch_bounds__(256) void dacl_normalize_f16(
    const float* __restrict__ orig, const float* __restrict__ aug,
    _Float16* __restrict__ oh, _Float16* __restrict__ ah) {
  const int gw   = blockIdx.x * 8 + (threadIdx.x >> 5);  // 0..8191
  const int lane = threadIdx.x & 31;

  const float* src;
  _Float16*    dst;
  int row;
  if (gw < BATCH) { src = orig; dst = oh; row = gw; }
  else            { src = aug;  dst = ah; row = gw - BATCH; }

  const float4* p = (const float4*)(src + (size_t)row * CHAN) + lane * 2;
  float4 x0 = p[0];
  float4 x1 = p[1];

  float ss = x0.x * x0.x + x0.y * x0.y + x0.z * x0.z + x0.w * x0.w +
             x1.x * x1.x + x1.y * x1.y + x1.z * x1.z + x1.w * x1.w;
#pragma unroll
  for (int off = 16; off >= 1; off >>= 1)
    ss += __shfl_xor(ss, off, 32);

  const float inv = 1.0f / fmaxf(sqrtf(ss), 1e-12f);

  v8h o;
  o[0] = (_Float16)(x0.x * inv); o[1] = (_Float16)(x0.y * inv);
  o[2] = (_Float16)(x0.z * inv); o[3] = (_Float16)(x0.w * inv);
  o[4] = (_Float16)(x1.x * inv); o[5] = (_Float16)(x1.y * inv);
  o[6] = (_Float16)(x1.z * inv); o[7] = (_Float16)(x1.w * inv);
  *(v8h*)(dst + (size_t)row * CHAN + lane * 8) = o;
}

// ---------------------------------------------------------------------------
// Kernel 2: zero the per-row partial accumulators and the scalar output.
// ---------------------------------------------------------------------------
__global__ __launch_bounds__(256) void dacl_zero(
    float* __restrict__ s_part, float* __restrict__ p_part,
    float* __restrict__ out) {
  const int i = blockIdx.x * 256 + threadIdx.x;
  if (i < BATCH) { s_part[i] = 0.0f; p_part[i] = 0.0f; }
  if (i == 0) out[0] = 0.0f;
}

// ---------------------------------------------------------------------------
// Kernel 3: fused GEMM + streaming fixed-shift logsumexp.
//
// Grid: (64 anchor groups, 4 key quarters). Block = 128 threads = 4 waves.
// Wave w owns the 16-anchor tile [bx*64 + w*16, +16); A fragments (16x256 f16)
// live in 64 VGPRs for the whole kernel.
// The block walks 128 key tiles (quarter by). Each 8 KB key tile is staged in
// LDS ONCE per block via async-to-LDS (double-buffered), then consumed by all
// 4 waves -> 4x less global traffic than per-wave streaming.
//
// Fragment layout convention (16-bit 16x32 A / 32x16 B, wave32):
//   lane l holds row (l & 15), K-halves [(l>>4)*16 .. +16) -> one v16h.
// C/D layout (ISA 7.12.2): VGPR v of lane l = element (row v + 8*(l>=16),
//   col l & 15) of the 16x16 f32 tile.
// ---------------------------------------------------------------------------
__global__ __launch_bounds__(128) void dacl_gemm_lse(
    const _Float16* __restrict__ oh, const _Float16* __restrict__ ah,
    float* __restrict__ s_part, float* __restrict__ p_part) {
  extern __shared__ char smem[];                 // 2 x TILE_BYTES

  const int lane    = threadIdx.x & 31;
  const int wave    = threadIdx.x >> 5;          // 0..3
  const int rowBase = blockIdx.x * 64 + wave * 16;

  const int fragRow   = lane & 15;               // row within a 16-row tile
  const int fragHalfB = (lane >> 4) * 32;        // byte offset of this lane's 16 halves

  // A fragments: this wave's 16 anchors, full K=256 (8 chunks of 32).
  v16h afrag[8];
#pragma unroll
  for (int kc = 0; kc < 8; ++kc)
    afrag[kc] = *(const v16h*)((const char*)oh +
                               (size_t)(rowBase + fragRow) * (CHAN * 2) +
                               kc * 64 + fragHalfB);

  float s[8], pos[8];
#pragma unroll
  for (int v = 0; v < 8; ++v) { s[v] = 0.0f; pos[v] = 0.0f; }

  const int accRowOff = (lane >> 4) * 8;         // C/D row offset for this lane
  const int colLane   = lane & 15;               // C/D column for this lane

  const int ntBeg = blockIdx.y * 128;            // key tiles: 0..255 oo, 256..511 oa
  const int ntEnd = ntBeg + 128;

  // Cooperative async stage of one 8 KB key tile into LDS buffer `buf`.
  // 128 threads x 64 B each, 4 x b128 per thread, tracked by ASYNCcnt.
  const int ldRow   = threadIdx.x >> 3;          // 0..15
  const int ldChunk = threadIdx.x & 7;           // 0..7 (64 B chunks)
  auto issue_tile = [&](int nt, int buf) {
    const int keyT = (nt >= 256) ? (nt - 256) : nt;
    const _Float16* kp = (nt >= 256) ? ah : oh;
    unsigned long long g = (unsigned long long)(uintptr_t)(
        (const char*)kp + (size_t)(keyT * 16 + ldRow) * (CHAN * 2) + ldChunk * 64);
    unsigned l = (unsigned)(buf * TILE_BYTES + ldRow * 512 + ldChunk * 64);
#pragma unroll
    for (int i = 0; i < 4; ++i) {
      asm volatile("global_load_async_to_lds_b128 %0, %1, off"
                   :: "v"(l + i * 16), "v"(g + (unsigned long long)(i * 16))
                   : "memory");
    }
  };

  issue_tile(ntBeg, 0);
  asm volatile("s_wait_asynccnt 0" ::: "memory");
  __syncthreads();

  for (int nt = ntBeg; nt < ntEnd; ++nt) {
    const int cur = (nt - ntBeg) & 1;
    if (nt + 1 < ntEnd) issue_tile(nt + 1, cur ^ 1);   // prefetch next tile

    const char* bt = smem + cur * TILE_BYTES;
    v8f acc = {};
#pragma unroll
    for (int kc = 0; kc < 8; ++kc) {
      v16h bfrag = *(const v16h*)(bt + fragRow * 512 + kc * 64 + fragHalfB);
      acc = __builtin_amdgcn_wmma_f32_16x16x32_f16(
          /*neg_a=*/false, afrag[kc], /*neg_b=*/false, bfrag,
          /*c_mod=*/(short)0, acc, /*reuse_a=*/false, /*reuse_b=*/false);
    }

    const bool isOA = (nt >= 256);
    const int  keyT = isOA ? (nt - 256) : nt;
    const int  colG = keyT * 16 + colLane;
#pragma unroll
    for (int v = 0; v < 8; ++v) {
      const int rG = rowBase + accRowOff + v;
      float x = acc[v] * INV_T;
      if (colG == rG) {
        if (isOA) pos[v] = x;          // positive pair (stays in denominator)
        else      x = -INFINITY;       // self-similarity: exp -> 0
      }
      s[v] += __expf(x - LSE_SHIFT);
    }

    asm volatile("s_wait_asynccnt 0" ::: "memory");    // prefetch landed
    __syncthreads();                                   // all waves done with cur
  }

  // Butterfly-reduce across the 16 lanes sharing each row (stays in half-wave).
#pragma unroll
  for (int off = 1; off < 16; off <<= 1) {
#pragma unroll
    for (int v = 0; v < 8; ++v) {
      s[v]   += __shfl_xor(s[v],   off, 32);
      pos[v] += __shfl_xor(pos[v], off, 32);
    }
  }

  // Fixed-shift partials are additive across the 4 key-quarter blocks.
  if ((lane & 15) == 0) {
#pragma unroll
    for (int v = 0; v < 8; ++v) {
      const int rG = rowBase + accRowOff + v;
      atomicAdd(&s_part[rG], s[v]);
      atomicAdd(&p_part[rG], pos[v]);
    }
  }
}

// ---------------------------------------------------------------------------
// Kernel 4: finalize  loss = mean_r( 10 + log(s_r) - pos_r )
// ---------------------------------------------------------------------------
__global__ __launch_bounds__(256) void dacl_finalize(
    const float* __restrict__ s_part, const float* __restrict__ p_part,
    float* __restrict__ out) {
  const int r = blockIdx.x * 256 + threadIdx.x;
  float c = (LSE_SHIFT + __logf(s_part[r]) - p_part[r]) * (1.0f / (float)BATCH);
  // reduce within wave first to cut atomics 32x
#pragma unroll
  for (int off = 16; off >= 1; off >>= 1)
    c += __shfl_xor(c, off, 32);
  if ((threadIdx.x & 31) == 0) atomicAdd(out, c);
}

// ---------------------------------------------------------------------------
// Host-side launch
// ---------------------------------------------------------------------------
extern "C" void kernel_launch(void* const* d_in, const int* in_sizes, int n_in,
                              void* d_out, int out_size, void* d_ws, size_t ws_size,
                              hipStream_t stream) {
  (void)in_sizes; (void)n_in; (void)out_size; (void)ws_size;

  const float* orig = (const float*)d_in[0];   // (4096, 256) f32
  const float* aug  = (const float*)d_in[1];   // (4096, 256) f32
  // d_in[2] = labels: unused by the loss math.
  float* out = (float*)d_out;

  const size_t featBytes = (size_t)BATCH * CHAN * 2;           // 2 MB each
  _Float16* oh     = (_Float16*)d_ws;
  _Float16* ah     = (_Float16*)((char*)d_ws + featBytes);
  float*    s_part = (float*)((char*)d_ws + 2 * featBytes);    // 16 KB
  float*    p_part = s_part + BATCH;                           // 16 KB

  dacl_normalize_f16<<<1024, 256, 0, stream>>>(orig, aug, oh, ah);
  dacl_zero<<<16, 256, 0, stream>>>(s_part, p_part, out);
  dacl_gemm_lse<<<dim3(64, 4), 128, 2 * TILE_BYTES, stream>>>(oh, ah, s_part, p_part);
  dacl_finalize<<<16, 256, 0, stream>>>(s_part, p_part, out);
}
